// RidgePrototypesEqx_46437186404599
// MI455X (gfx1250) — matrix operand order repaired and checked by reading, not compiled, not run on hardware
//
#include <hip/hip_runtime.h>

// ---- CDNA5 (gfx1250) wave32 WMMA types ----
typedef __bf16 v16bf __attribute__((ext_vector_type(16)));
typedef __bf16 v8bf  __attribute__((ext_vector_type(8)));
typedef __bf16 v4bf  __attribute__((ext_vector_type(4)));
typedef float  v8f   __attribute__((ext_vector_type(8)));
typedef float  v4f   __attribute__((ext_vector_type(4)));

#define D_DIM      64
#define K_PROT     256
#define M_TILE     128
#define LDS_STRIDE 80      // bf16 per row: 64 data + 16 pad -> 160B rows, bank-rotating, 32B aligned
#define NTHREADS   256     // 8 wave32
#define GRID_CAP   1024    // persistent blocks; mus staged once per block

static __device__ __forceinline__ v16bf cat8(v8bf a, v8bf b) {
  return __builtin_shufflevector(a, b, 0,1,2,3,4,5,6,7,8,9,10,11,12,13,14,15);
}

__global__ __launch_bounds__(NTHREADS)
void proto_assign_kernel(const float* __restrict__ X,
                         const float* __restrict__ mus,
                         int* __restrict__ out, int n, int n_tiles)
{
  extern __shared__ char smem_raw[];
  __bf16* mu_hi = (__bf16*)smem_raw;                       // [256][80]
  __bf16* mu_lo = mu_hi + K_PROT * LDS_STRIDE;             // [256][80]
  __bf16* x_hi  = mu_lo + K_PROT * LDS_STRIDE;             // [128][80]
  __bf16* x_lo  = x_hi  + M_TILE * LDS_STRIDE;             // [128][80]
  float*  mu_sq = (float*)(x_lo + M_TILE * LDS_STRIDE);    // [256] f32 (exact)

  const int tid  = threadIdx.x;
  const int lane = tid & 31;
  const int wave = tid >> 5;
  const int half = lane >> 4;     // 0: lanes 0-15, 1: lanes 16-31
  const int ncol = lane & 15;

  // ================= one-time prologue: stage mus (hi+lo) and mu_sq =================
  {
    const v4f* src = (const v4f*)mus;
    for (int i = tid; i < K_PROT * D_DIM / 4; i += NTHREADS) {
      v4f v = src[i];
      int row = i >> 4;           // 16 float4 per 64-elem row
      int c4  = i & 15;
      __bf16 h0 = (__bf16)v.x, h1 = (__bf16)v.y, h2 = (__bf16)v.z, h3 = (__bf16)v.w;
      v4bf hv = { h0, h1, h2, h3 };
      v4bf lv = { (__bf16)(v.x - (float)h0), (__bf16)(v.y - (float)h1),
                  (__bf16)(v.z - (float)h2), (__bf16)(v.w - (float)h3) };
      int off = row * LDS_STRIDE + c4 * 4;
      *(v4bf*)(mu_hi + off) = hv;
      *(v4bf*)(mu_lo + off) = lv;
    }
  }
  {
    const v4f* mrow = (const v4f*)(mus + tid * D_DIM);
    float s = 0.f;
#pragma unroll
    for (int j = 0; j < D_DIM / 4; ++j) {
      v4f v = mrow[j];
      s = fmaf(v.x, v.x, s); s = fmaf(v.y, v.y, s);
      s = fmaf(v.z, v.z, s); s = fmaf(v.w, v.w, s);
    }
    mu_sq[tid] = s;
  }
  __syncthreads();

  // ================= persistent grid-stride loop over M-tiles =================
  for (int tile = blockIdx.x; tile < n_tiles; tile += gridDim.x) {
    const long long m_block = (long long)tile * M_TILE;

    // ---- stage X tile (streamed once: non-temporal; zero-fill rows beyond n) ----
    {
      const v4f* src = (const v4f*)X;
      const long long base = m_block * (D_DIM / 4);
      const long long lim  = (long long)n * (D_DIM / 4);
      for (int i = tid; i < M_TILE * D_DIM / 4; i += NTHREADS) {
        long long g = base + i;
        v4f v = { 0.f, 0.f, 0.f, 0.f };
        if (g < lim) v = __builtin_nontemporal_load(&src[g]);
        int row = i >> 4;
        int c4  = i & 15;
        __bf16 h0 = (__bf16)v.x, h1 = (__bf16)v.y, h2 = (__bf16)v.z, h3 = (__bf16)v.w;
        v4bf hv = { h0, h1, h2, h3 };
        v4bf lv = { (__bf16)(v.x - (float)h0), (__bf16)(v.y - (float)h1),
                    (__bf16)(v.z - (float)h2), (__bf16)(v.w - (float)h3) };
        int off = row * LDS_STRIDE + c4 * 4;
        *(v4bf*)(x_hi + off) = hv;
        *(v4bf*)(x_lo + off) = lv;
      }
    }
    __syncthreads();

    // ---- preload A fragments (this wave's 16 rows), 2 K-chunks of 32, hi+lo ----
    // A 16x32, contiguous k order chosen here (B uses the same): lane holds row m=lane&15; elem e <-> K=16*half+e
    //  => one contiguous 32B run per chunk (matches B).
    const int rowx = wave * 16 + ncol;
    v16bf ah[2], al[2];
#pragma unroll
    for (int c = 0; c < 2; ++c) {
      const __bf16* ph = x_hi + rowx * LDS_STRIDE + 32 * c + 16 * half;
      const __bf16* pl = x_lo + rowx * LDS_STRIDE + 32 * c + 16 * half;
      ah[c] = cat8(*(const v8bf*)ph, *(const v8bf*)(ph + 8));
      al[c] = cat8(*(const v8bf*)pl, *(const v8bf*)(pl + 8));
    }

    float best_val[8];
    int   best_idx[8];
#pragma unroll
    for (int r = 0; r < 8; ++r) { best_val[r] = 3.402823466e38f; best_idx[r] = K_PROT; }

    // ---- 16 K-tiles of 16 protos; two concurrent accumulator chains ----
    for (int t = 0; t < K_PROT / 16; t += 2) {
      v8f acc0 = {};
      v8f acc1 = {};
      // B 32x16, same contiguous k order as A: lane holds col N=lane&15; elem e <-> K = 16*half + e
      // => one contiguous 32B run per chunk (two b128 LDS loads).
      const __bf16* bh0_row = mu_hi + ((t    ) * 16 + ncol) * LDS_STRIDE + 16 * half;
      const __bf16* bl0_row = mu_lo + ((t    ) * 16 + ncol) * LDS_STRIDE + 16 * half;
      const __bf16* bh1_row = mu_hi + ((t + 1) * 16 + ncol) * LDS_STRIDE + 16 * half;
      const __bf16* bl1_row = mu_lo + ((t + 1) * 16 + ncol) * LDS_STRIDE + 16 * half;
#pragma unroll
      for (int c = 0; c < 2; ++c) {
        v16bf bh_0 = cat8(*(const v8bf*)(bh0_row + 32 * c), *(const v8bf*)(bh0_row + 32 * c + 8));
        v16bf bl_0 = cat8(*(const v8bf*)(bl0_row + 32 * c), *(const v8bf*)(bl0_row + 32 * c + 8));
        v16bf bh_1 = cat8(*(const v8bf*)(bh1_row + 32 * c), *(const v8bf*)(bh1_row + 32 * c + 8));
        v16bf bl_1 = cat8(*(const v8bf*)(bl1_row + 32 * c), *(const v8bf*)(bl1_row + 32 * c + 8));
        // split-precision product: Xh*Mh + Xh*Ml + Xl*Mh  (~f32 accuracy)
        acc0 = __builtin_amdgcn_wmma_f32_16x16x32_bf16(false, ah[c], false, bh_0, (short)0, acc0, false, false);
        acc1 = __builtin_amdgcn_wmma_f32_16x16x32_bf16(false, ah[c], false, bh_1, (short)0, acc1, false, false);
        acc0 = __builtin_amdgcn_wmma_f32_16x16x32_bf16(false, ah[c], false, bl_0, (short)0, acc0, false, false);
        acc1 = __builtin_amdgcn_wmma_f32_16x16x32_bf16(false, ah[c], false, bl_1, (short)0, acc1, false, false);
        acc0 = __builtin_amdgcn_wmma_f32_16x16x32_bf16(false, al[c], false, bh_0, (short)0, acc0, false, false);
        acc1 = __builtin_amdgcn_wmma_f32_16x16x32_bf16(false, al[c], false, bh_1, (short)0, acc1, false, false);
      }
      // score = ||mu||^2 - 2*<x,mu>  (drop constant ||x||^2; ordering preserved).
      // Strict '<' keeps the lowest index per lane (indices strictly increase).
      float msq0 = mu_sq[(t    ) * 16 + ncol];
      float msq1 = mu_sq[(t + 1) * 16 + ncol];
      int idx0 = (t    ) * 16 + ncol;
      int idx1 = (t + 1) * 16 + ncol;
#pragma unroll
      for (int r = 0; r < 8; ++r) {
        float s0 = fmaf(-2.f, acc0[r], msq0);
        float s1 = fmaf(-2.f, acc1[r], msq1);
        if (s0 < best_val[r]) { best_val[r] = s0; best_idx[r] = idx0; }
        if (s1 < best_val[r]) { best_val[r] = s1; best_idx[r] = idx1; }
      }
    }

    // ---- argmin reduction across the 16 lanes holding each row ----
    // (xor 1,2,4,8 stays within each 16-lane half; full tie-break here only)
#pragma unroll
    for (int r = 0; r < 8; ++r) {
      float v = best_val[r]; int ix = best_idx[r];
#pragma unroll
      for (int m = 1; m <= 8; m <<= 1) {
        float ov = __shfl_xor(v, m, 32);
        int   oi = __shfl_xor(ix, m, 32);
        if (ov < v || (ov == v && oi < ix)) { v = ov; ix = oi; }
      }
      best_idx[r] = ix;
    }

    // C/D layout: VGPR r, lanes 0-15 -> row r; lanes 16-31 -> row r+8.
    if (ncol == 0) {
      long long m0 = m_block + wave * 16 + half * 8;
#pragma unroll
      for (int r = 0; r < 8; ++r) {
        long long m = m0 + r;
        if (m < n) __builtin_nontemporal_store(best_idx[r], &out[m]);
      }
    }
    __syncthreads();   // protect x_hi/x_lo before next tile restage
  }
}

extern "C" void kernel_launch(void* const* d_in, const int* in_sizes, int n_in,
                              void* d_out, int out_size, void* d_ws, size_t ws_size,
                              hipStream_t stream) {
  const float* X   = (const float*)d_in[0];   // [N, 64] f32
  const float* mus = (const float*)d_in[1];   // [256, 64] f32
  int* out = (int*)d_out;                     // [N] int32 argmin indices
  const int n = in_sizes[0] / D_DIM;
  const int n_tiles = (n + M_TILE - 1) / M_TILE;
  const int blocks = n_tiles < GRID_CAP ? n_tiles : GRID_CAP;
  const size_t shmem =
      (size_t)(2 * K_PROT * LDS_STRIDE + 2 * M_TILE * LDS_STRIDE) * 2 /*bf16*/ +
      (size_t)K_PROT * sizeof(float);         // ~124 KB, fits CDNA5 320KB/WGP LDS
  proto_assign_kernel<<<blocks, NTHREADS, shmem, stream>>>(X, mus, out, n, n_tiles);
}
